// BidirVimMamba2Encoder_50955491999885
// MI455X (gfx1250) — compile-verified, matrix operand order checked by reading, not run on hardware
//
#include <hip/hip_runtime.h>
#include <hip/hip_bf16.h>

// ---------------------------------------------------------------------------
// BidirVimMamba2Encoder for MI455X (gfx1250, wave32, WMMA)
//
// GEMMs (in_proj / out_proj / FFN1 / FFN2) run on v_wmma_f32_16x16x32_f16
// with fp16 inputs and f32 accumulation. Weights are converted to transposed
// fp16 once per launch so both A and B fragments are contiguous 16B loads
// matching the CDNA5 16-bit 16x32 fragment layout. The SSM scan keeps the
// full (P=64, N=64) state in registers: 64 blocks (one per (batch, head)),
// 256 threads, 16 state floats per thread, 4-lane shuffle reduction.
// ---------------------------------------------------------------------------

#define BQ    4
#define TQ    1024
#define DMOD  512
#define DI    1024      // D_INNER
#define NH    16        // heads
#define PPD   64        // head dim
#define NNS   64        // d_state
#define CONVD 1152      // D_INNER + 2*N
#define DIP   2192      // 2*DI + 2*N + H
#define FFND  2048
#define BT    (BQ * TQ) // 4096
#define EPSF  1e-5f

typedef _Float16 v16h __attribute__((ext_vector_type(16)));
typedef _Float16 h8   __attribute__((ext_vector_type(8)));
typedef float    v8f  __attribute__((ext_vector_type(8)));

// ---------------------------------------------------------------- helpers
__device__ __forceinline__ float silu_f(float x) {
    return x / (1.0f + __expf(-x));
}
__device__ __forceinline__ float gelu_f(float x) {
    const float c = 0.7978845608028654f; // sqrt(2/pi)
    float x3 = x * x * x;
    return 0.5f * x * (1.0f + tanhf(c * (x + 0.044715f * x3)));
}

// --------------------------------------------------------------- pos enc
__global__ void pos_enc_k(const float* __restrict__ x, float* __restrict__ y) {
    int idx = blockIdx.x * 256 + threadIdx.x;
    if (idx >= BT * DMOD) return;
    int d = idx % DMOD;
    int t = (idx / DMOD) % TQ;
    int i2 = d >> 1;
    float ang = (float)t * __expf(-logf(10000.0f) * (float)(2 * i2) / (float)DMOD);
    float pe = (d & 1) ? cosf(ang) : sinf(ang);
    y[idx] = x[idx] + pe;
}

// -------------------------------------------------------- weight convert
// W: [L2][Kd][Nd] f32  ->  WT: [L2][Nd][Kd] fp16
__global__ void wt_k(const float* __restrict__ W, _Float16* __restrict__ WT,
                     int Kd, int Nd) {
    long idx = (long)blockIdx.x * 256 + threadIdx.x;
    long per = (long)Kd * Nd;
    long total = 4 * per;
    if (idx >= total) return;
    long l = idx / per;
    long rem = idx % per;
    long k = rem / Nd, n = rem % Nd;
    WT[l * per + n * (long)Kd + k] = (_Float16)W[idx];
}

// ------------------------------------------------------------- layernorm
// One block per row of DMOD=512; emits fp16 (GEMM A input).
__global__ void layernorm_k(const float* __restrict__ x,
                            const float* __restrict__ w,
                            const float* __restrict__ b,
                            _Float16* __restrict__ out) {
    __shared__ float s1[256], s2[256];
    int row = blockIdx.x;
    const float* xr = x + (size_t)row * DMOD;
    float v0 = xr[threadIdx.x];
    float v1 = xr[threadIdx.x + 256];
    s1[threadIdx.x] = v0 + v1;
    s2[threadIdx.x] = v0 * v0 + v1 * v1;
    __syncthreads();
    for (int s = 128; s > 0; s >>= 1) {
        if (threadIdx.x < s) {
            s1[threadIdx.x] += s1[threadIdx.x + s];
            s2[threadIdx.x] += s2[threadIdx.x + s];
        }
        __syncthreads();
    }
    float mean = s1[0] * (1.0f / DMOD);
    float var  = s2[0] * (1.0f / DMOD) - mean * mean;
    float rstd = rsqrtf(var + EPSF);
    _Float16* o = out + (size_t)row * DMOD;
    o[threadIdx.x]       = (_Float16)((v0 - mean) * rstd * w[threadIdx.x] + b[threadIdx.x]);
    o[threadIdx.x + 256] = (_Float16)((v1 - mean) * rstd * w[threadIdx.x + 256] + b[threadIdx.x + 256]);
}

// ------------------------------------------------------------ WMMA GEMM
// C[M=4096][Nout] = A[M][K](fp16) * WT[Nout][K](fp16)^T  (+bias +residual,
// optional GELU). One 16x16 tile per wave, 8 waves per block.
// Fragment layout per CDNA5 ISA, 16-bit A 16x32:
//   lane 0-15 : M=lane,   K = k0+0..7   and  k0+16..23
//   lane16-31 : M=lane-16,K = k0+8..15  and  k0+24..31
// B fragment is the same layout over WT rows (N columns of W).
__device__ __forceinline__ v16h load_frag(const _Float16* __restrict__ base,
                                          int ld, int k0) {
    int lane = threadIdx.x & 31;
    int half = lane >> 4;
    int r    = lane & 15;
    const _Float16* p = base + (size_t)r * ld + k0 + half * 8;
    h8 lo = *(const h8*)(p);
    h8 hi = *(const h8*)(p + 16);
    v16h f;
#pragma unroll
    for (int i = 0; i < 8; ++i) { f[i] = lo[i]; f[i + 8] = hi[i]; }
    return f;
}

__global__ void gemm_k(const _Float16* __restrict__ A,
                       const _Float16* __restrict__ WT,
                       const float* __restrict__ bias,
                       const float* __restrict__ resid,
                       float* __restrict__ outf,
                       _Float16* __restrict__ outh,
                       int K, int Nout, int act) {
    int wave = threadIdx.x >> 5;
    int tn = blockIdx.x * 8 + wave;
    if (tn * 16 >= Nout) return;              // wave-uniform: EXEC stays full
    int m0 = blockIdx.y * 16;
    int n0 = tn * 16;
    const _Float16* Abase = A  + (size_t)m0 * K;
    const _Float16* Bbase = WT + (size_t)n0 * K;

    v8f acc = {0.f, 0.f, 0.f, 0.f, 0.f, 0.f, 0.f, 0.f};
    for (int k0 = 0; k0 < K; k0 += 32) {
        v16h a = load_frag(Abase, K, k0);
        v16h b = load_frag(Bbase, K, k0);
        acc = __builtin_amdgcn_wmma_f32_16x16x32_f16(
            false, a, false, b, (short)0, acc, false, false);
    }

    int lane = threadIdx.x & 31;
    int half = lane >> 4;
    int col  = lane & 15;
    int n = n0 + col;
    float bv = bias ? bias[n] : 0.0f;
#pragma unroll
    for (int r = 0; r < 8; ++r) {
        int m = m0 + r + (half ? 8 : 0);
        float v = acc[r] + bv;
        if (resid) v += resid[(size_t)m * Nout + n];
        if (act) v = gelu_f(v);
        if (outf) outf[(size_t)m * Nout + n] = v;
        if (outh) outh[(size_t)m * Nout + n] = (_Float16)v;
    }
}

// ------------------------------------------------------- causal conv+SiLU
// reads xBC slice of zxbcdt ([row][1024 + c]), writes xBCout [row][1152]
__global__ void conv_k(const float* __restrict__ zx,
                       const float* __restrict__ cw,
                       const float* __restrict__ cb,
                       float* __restrict__ out) {
    int idx = blockIdx.x * 256 + threadIdx.x;
    if (idx >= BT * CONVD) return;
    int c = idx % CONVD;
    int row = idx / CONVD;
    int t = row % TQ;
    int b = row / TQ;
    float acc = cb[c];
#pragma unroll
    for (int k = 0; k < 4; ++k) {
        int tt = t + k - 3;
        if (tt >= 0)
            acc += cw[k * CONVD + c] * zx[((size_t)(b * TQ + tt)) * DIP + DI + c];
    }
    out[(size_t)row * CONVD + c] = silu_f(acc);
}

// ---------------------------------------------------------- dt / dA prep
__global__ void dt_k(const float* __restrict__ zx,
                     const float* __restrict__ dtb,
                     const float* __restrict__ Alog,
                     float* __restrict__ dt, float* __restrict__ dA) {
    int idx = blockIdx.x * 256 + threadIdx.x;
    if (idx >= BT * NH) return;
    int h = idx % NH;
    int row = idx / NH;
    float x = zx[(size_t)row * DIP + DI + CONVD + h] + dtb[h];
    float sp = (x > 20.0f) ? x : log1pf(__expf(x));
    dt[idx] = sp;
    dA[idx] = __expf(-__expf(Alog[h]) * sp);
}

// --------------------------------------------------------------- SSM scan
// one block per (b,h); thread owns p = tid>>2, n in [ (tid&3)*16, +16 )
__global__ void scan_k(const float* __restrict__ xBC,
                       const float* __restrict__ dt,
                       const float* __restrict__ dA,
                       float* __restrict__ ys) {
    int b = blockIdx.x / NH;
    int h = blockIdx.x % NH;
    int tid = threadIdx.x;
    int p  = tid >> 2;
    int n0 = (tid & 3) * 16;

    float st[16];
#pragma unroll
    for (int i = 0; i < 16; ++i) st[i] = 0.0f;

    for (int t = 0; t < TQ; ++t) {
        int row = b * TQ + t;
        const float* xr = xBC + (size_t)row * CONVD;
        float xt  = xr[h * PPD + p];
        float dtv = dt[row * NH + h];
        float dAv = dA[row * NH + h];
        float coef = dtv * xt;
        const float* Bp = xr + DI + n0;
        const float* Cp = xr + DI + NNS + n0;
        float part = 0.0f;
#pragma unroll
        for (int i = 0; i < 16; ++i) {
            st[i] = st[i] * dAv + coef * Bp[i];
            part += st[i] * Cp[i];
        }
        part += __shfl_xor(part, 1, 32);
        part += __shfl_xor(part, 2, 32);
        if ((tid & 3) == 0)
            ys[(size_t)row * DI + h * PPD + p] = part;
    }
}

// ------------------------------------------------- gated RMSNorm (-> fp16)
// u = (ys + Dp[h]*xs) * silu(z); out = rmsnorm(u) * gw
__global__ void rmsgate_k(const float* __restrict__ ys,
                          const float* __restrict__ xBC,
                          const float* __restrict__ zx,
                          const float* __restrict__ Dp,
                          const float* __restrict__ gw,
                          _Float16* __restrict__ out) {
    __shared__ float s1[256];
    int row = blockIdx.x;
    float u[4];
    float acc = 0.0f;
#pragma unroll
    for (int j = 0; j < 4; ++j) {
        int i = threadIdx.x * 4 + j;      // 0..1023
        int h = i >> 6;
        float xv = xBC[(size_t)row * CONVD + i];
        float yv = ys[(size_t)row * DI + i] + Dp[h] * xv;
        float zv = zx[(size_t)row * DIP + i];
        float uu = yv * silu_f(zv);
        u[j] = uu;
        acc += uu * uu;
    }
    s1[threadIdx.x] = acc;
    __syncthreads();
    for (int s = 128; s > 0; s >>= 1) {
        if (threadIdx.x < s) s1[threadIdx.x] += s1[threadIdx.x + s];
        __syncthreads();
    }
    float rms = rsqrtf(s1[0] * (1.0f / DI) + EPSF);
#pragma unroll
    for (int j = 0; j < 4; ++j) {
        int i = threadIdx.x * 4 + j;
        out[(size_t)row * DI + i] = (_Float16)(u[j] * rms * gw[i]);
    }
}

// ------------------------------------------------ reverse / combine / mask
__global__ void reverse_k(const float* __restrict__ x, float* __restrict__ y) {
    int idx = blockIdx.x * 256 + threadIdx.x;
    if (idx >= BT * DMOD) return;
    int d = idx % DMOD;
    int row = idx / DMOD;
    int t = row % TQ, b = row / TQ;
    y[idx] = x[((size_t)(b * TQ + (TQ - 1 - t))) * DMOD + d];
}

__global__ void combine_k(const float* __restrict__ pf,
                          const float* __restrict__ pb,
                          float* __restrict__ x) {
    int idx = blockIdx.x * 256 + threadIdx.x;
    if (idx >= BT * DMOD) return;
    int d = idx % DMOD;
    int row = idx / DMOD;
    int t = row % TQ, b = row / TQ;
    float vb = pb[((size_t)(b * TQ + (TQ - 1 - t))) * DMOD + d];
    x[idx] = 0.5f * (pf[idx] + vb);
}

__global__ void mask_k(const float* __restrict__ x,
                       const int* __restrict__ lengths,
                       float* __restrict__ out) {
    int idx = blockIdx.x * 256 + threadIdx.x;
    if (idx >= BT * DMOD) return;
    int row = idx / DMOD;
    int t = row % TQ, b = row / TQ;
    out[idx] = (t < lengths[b]) ? x[idx] : 0.0f;
}

// ===========================================================================
extern "C" void kernel_launch(void* const* d_in, const int* in_sizes, int n_in,
                              void* d_out, int out_size, void* d_ws, size_t ws_size,
                              hipStream_t stream) {
    const float* x_in  = (const float*)d_in[0];
    const int*   lens  = (const int*)d_in[1];
    const float* ln1w  = (const float*)d_in[2];
    const float* ln1b  = (const float*)d_in[3];
    const float* ipw   = (const float*)d_in[4];
    const float* cw    = (const float*)d_in[5];
    const float* cb    = (const float*)d_in[6];
    const float* dtb   = (const float*)d_in[7];
    const float* Alog  = (const float*)d_in[8];
    const float* Dp    = (const float*)d_in[9];
    const float* gw    = (const float*)d_in[10];
    const float* opw   = (const float*)d_in[11];
    const float* ln2w  = (const float*)d_in[12];
    const float* ln2b  = (const float*)d_in[13];
    const float* w1    = (const float*)d_in[14];
    const float* b1    = (const float*)d_in[15];
    const float* w2    = (const float*)d_in[16];
    const float* b2    = (const float*)d_in[17];

    char* ws = (char*)d_ws;
    size_t off = 0;
    auto carve = [&](size_t bytes) -> char* {
        char* p = ws + off;
        off = (off + bytes + 255) & ~(size_t)255;
        return p;
    };

    // fp16 transposed weights (per layer-block l = 0..3)
    _Float16* ipwT = (_Float16*)carve((size_t)4 * DIP  * DMOD * sizeof(_Float16));
    _Float16* opwT = (_Float16*)carve((size_t)4 * DMOD * DI   * sizeof(_Float16));
    _Float16* w1T  = (_Float16*)carve((size_t)4 * FFND * DMOD * sizeof(_Float16));
    _Float16* w2T  = (_Float16*)carve((size_t)4 * DMOD * FFND * sizeof(_Float16));
    // activations
    float*    xcur = (float*)carve((size_t)BT * DMOD * sizeof(float));
    float*    xrev = (float*)carve((size_t)BT * DMOD * sizeof(float));
    float*    pfwd = (float*)carve((size_t)BT * DMOD * sizeof(float));
    float*    pbwd = (float*)carve((size_t)BT * DMOD * sizeof(float));
    float*    xmid = (float*)carve((size_t)BT * DMOD * sizeof(float));
    _Float16* hbuf = (_Float16*)carve((size_t)BT * DMOD * sizeof(_Float16));
    float*    zx   = (float*)carve((size_t)BT * DIP   * sizeof(float));
    float*    xBCb = (float*)carve((size_t)BT * CONVD * sizeof(float));
    float*    dtbf = (float*)carve((size_t)BT * NH * sizeof(float));
    float*    dAbf = (float*)carve((size_t)BT * NH * sizeof(float));
    float*    ysb  = (float*)carve((size_t)BT * DI * sizeof(float));
    _Float16* ubuf = (_Float16*)carve((size_t)BT * DI * sizeof(_Float16));
    _Float16* a1b  = (_Float16*)carve((size_t)BT * FFND * sizeof(_Float16));
    (void)ws_size; (void)in_sizes; (void)n_in; (void)out_size;

    // -- weight conversion (every launch; deterministic, no cached state) --
    {
        long n;
        n = 4L * DMOD * DIP;
        wt_k<<<(int)((n + 255) / 256), 256, 0, stream>>>(ipw, ipwT, DMOD, DIP);
        n = 4L * DI * DMOD;
        wt_k<<<(int)((n + 255) / 256), 256, 0, stream>>>(opw, opwT, DI, DMOD);
        n = 4L * DMOD * FFND;
        wt_k<<<(int)((n + 255) / 256), 256, 0, stream>>>(w1, w1T, DMOD, FFND);
        n = 4L * FFND * DMOD;
        wt_k<<<(int)((n + 255) / 256), 256, 0, stream>>>(w2, w2T, FFND, DMOD);
    }

    const int MT = BT / 16;                 // 256 row tiles
    auto run_block = [&](const float* xin, float* pout, int l) {
        // LN1 -> fp16
        layernorm_k<<<BT, 256, 0, stream>>>(xin, ln1w + l * DMOD, ln1b + l * DMOD, hbuf);
        // in_proj: [4096 x 512] x [512 x 2192]
        {
            dim3 g((DIP / 16 + 7) / 8, MT);
            gemm_k<<<g, 256, 0, stream>>>(hbuf, ipwT + (size_t)l * DIP * DMOD,
                                          nullptr, nullptr, zx, nullptr,
                                          DMOD, DIP, 0);
        }
        // causal conv + SiLU
        conv_k<<<(BT * CONVD + 255) / 256, 256, 0, stream>>>(
            zx, cw + (size_t)l * 4 * CONVD, cb + (size_t)l * CONVD, xBCb);
        // dt = softplus(dtr + bias), dA = exp(-exp(Alog)*dt)
        dt_k<<<(BT * NH + 255) / 256, 256, 0, stream>>>(
            zx, dtb + l * NH, Alog + l * NH, dtbf, dAbf);
        // sequential SSM scan
        scan_k<<<BQ * NH, 256, 0, stream>>>(xBCb, dtbf, dAbf, ysb);
        // y = ys + Dp*x; gated rmsnorm -> fp16
        rmsgate_k<<<BT, 256, 0, stream>>>(ysb, xBCb, zx, Dp + l * NH,
                                          gw + (size_t)l * DI, ubuf);
        // out_proj + residual(xin) -> xmid
        {
            dim3 g((DMOD / 16 + 7) / 8, MT);
            gemm_k<<<g, 256, 0, stream>>>(ubuf, opwT + (size_t)l * DMOD * DI,
                                          nullptr, xin, xmid, nullptr,
                                          DI, DMOD, 0);
        }
        // LN2 -> fp16
        layernorm_k<<<BT, 256, 0, stream>>>(xmid, ln2w + l * DMOD, ln2b + l * DMOD, hbuf);
        // FFN1 + bias + GELU -> fp16
        {
            dim3 g((FFND / 16 + 7) / 8, MT);
            gemm_k<<<g, 256, 0, stream>>>(hbuf, w1T + (size_t)l * FFND * DMOD,
                                          b1 + (size_t)l * FFND, nullptr,
                                          nullptr, a1b, DMOD, FFND, 1);
        }
        // FFN2 + bias + residual(xmid) -> pout
        {
            dim3 g((DMOD / 16 + 7) / 8, MT);
            gemm_k<<<g, 256, 0, stream>>>(a1b, w2T + (size_t)l * DMOD * FFND,
                                          b2 + l * DMOD, xmid, pout, nullptr,
                                          FFND, DMOD, 0);
        }
    };

    const int NE = (BT * DMOD + 255) / 256;
    pos_enc_k<<<NE, 256, 0, stream>>>(x_in, xcur);

    for (int i = 0; i < 2; ++i) {
        run_block(xcur, pfwd, 2 * i);
        reverse_k<<<NE, 256, 0, stream>>>(xcur, xrev);
        run_block(xrev, pbwd, 2 * i + 1);
        combine_k<<<NE, 256, 0, stream>>>(pfwd, pbwd, xcur);
    }

    mask_k<<<NE, 256, 0, stream>>>(xcur, lens, (float*)d_out);
}
